// CrossAttentionT2S_2078764171647
// MI455X (gfx1250) — matrix operand order repaired and checked by reading, not compiled, not run on hardware
//
#include <hip/hip_runtime.h>

#define DIMD   768
#define HEADS  12
#define HDIM   64
#define TT     8
#define NTOK   196
#define BB     16
#define BN     (BB * NTOK)      // 3136
#define MROWS  (BN * TT)        // 25088
#define KDIM   768

typedef __attribute__((ext_vector_type(16))) _Float16 v16h;
typedef __attribute__((ext_vector_type(8)))  _Float16 v8h;
typedef __attribute__((ext_vector_type(8)))  float    v8f;

// ---------------------------------------------------------------------------
// Prep: fused rearrange + time-pos add, f32 -> f16, plus output-row scatter map
// ---------------------------------------------------------------------------
__global__ void prep_kernel(const float* __restrict__ s_x,
                            const float* __restrict__ t_x,
                            const float* __restrict__ ctp,
                            const float* __restrict__ vtp,
                            _Float16* __restrict__ qin,
                            _Float16* __restrict__ kvin,
                            int* __restrict__ rowmap)
{
    size_t idx = (size_t)blockIdx.x * blockDim.x + threadIdx.x;
    if (idx >= (size_t)MROWS * DIMD) return;
    int d  = (int)(idx % DIMD);
    int m  = (int)(idx / DIMD);
    int tt = m % TT;
    int bn = m / TT;
    int nn = bn % NTOK;
    int b  = bn / NTOK;

    size_t srow = ((size_t)(1 + nn) * (BB * TT) + (size_t)b * TT + tt) * DIMD;
    qin[idx] = (_Float16)(s_x[srow + d] + ctp[tt * DIMD + d]);

    size_t trow = ((size_t)b * (TT * NTOK) + (size_t)tt * NTOK + nn) * DIMD;
    kvin[idx] = (_Float16)(t_x[trow + d] + vtp[tt * DIMD + d]);

    if (d == 0) rowmap[m] = (int)srow;
}

__global__ void f32_to_f16_kernel(const float* __restrict__ src,
                                  _Float16* __restrict__ dst, int n)
{
    int i = blockIdx.x * blockDim.x + threadIdx.x;
    if (i < n) dst[i] = (_Float16)src[i];
}

__global__ void copy_cls_kernel(const float* __restrict__ s_x,
                                float* __restrict__ out)
{
    int i = blockIdx.x * blockDim.x + threadIdx.x;
    if (i < BB * TT * DIMD) out[i] = s_x[i];
}

// ---------------------------------------------------------------------------
// WMMA fragment helpers
// ---------------------------------------------------------------------------
struct Frags { v16h a0, a1, b0, b1, b2, b3; };

__device__ __forceinline__ Frags load_frags(const _Float16* __restrict__ x0,
                                            const _Float16* __restrict__ x1,
                                            const _Float16* __restrict__ w0,
                                            int k0)
{
    Frags f;
    union { v16h v; v8h h[2]; } u;
    u.h[0] = *(const v8h*)(x0 + k0);
    u.h[1] = *(const v8h*)(x0 + k0 + 8);
    f.a0 = u.v;
    u.h[0] = *(const v8h*)(x1 + k0);
    u.h[1] = *(const v8h*)(x1 + k0 + 8);
    f.a1 = u.v;
    const _Float16* wp = w0 + k0;
    f.b0 = *(const v16h*)(wp);
    f.b1 = *(const v16h*)(wp + 16 * (size_t)KDIM);
    f.b2 = *(const v16h*)(wp + 32 * (size_t)KDIM);
    f.b3 = *(const v16h*)(wp + 48 * (size_t)KDIM);
    return f;
}

#define WMMA_F16(A, B, C) \
    __builtin_amdgcn_wmma_f32_16x16x32_f16(false, (A), false, (B), (short)0, (C), false, false)

__device__ __forceinline__ void do_wmma8(v8f acc[8], const Frags& f)
{
    acc[0] = WMMA_F16(f.a0, f.b0, acc[0]);
    acc[1] = WMMA_F16(f.a0, f.b1, acc[1]);
    acc[2] = WMMA_F16(f.a0, f.b2, acc[2]);
    acc[3] = WMMA_F16(f.a0, f.b3, acc[3]);
    acc[4] = WMMA_F16(f.a1, f.b0, acc[4]);
    acc[5] = WMMA_F16(f.a1, f.b1, acc[5]);
    acc[6] = WMMA_F16(f.a1, f.b2, acc[6]);
    acc[7] = WMMA_F16(f.a1, f.b3, acc[7]);
}

// ---------------------------------------------------------------------------
// WMMA GEMM: C[M,NTOT] = X[M,K] * W[NTOT,K]^T + bias, K = 768, f16 in, f32 acc.
// One wave computes a 32(M) x 64(N) strip (8 accumulators, 2 A frags x 4 B
// frags). All NTOT/64 n-tiles of a 32-row m-strip live in one workgroup so
// the A-fragment re-reads hit the WGP cache. Two-stage software pipeline
// (ping-pong fragment registers) so loads for stage k+32 overlap WMMAs of
// stage k instead of stalling at s_wait_loadcnt 0.
// ---------------------------------------------------------------------------
template<int NTOT, bool HALF_OUT>
__global__ __launch_bounds__(32 * (NTOT / 64)) void gemm_wmma_kernel(
    const _Float16* __restrict__ X,
    const _Float16* __restrict__ W,
    const float*    __restrict__ bias,
    _Float16*       __restrict__ outH,
    float*          __restrict__ outF,
    const int*      __restrict__ rowmap)
{
    int lane = threadIdx.x & 31;
    int nt   = threadIdx.x >> 5;        // n-tile within the block
    int mt   = blockIdx.x;              // 32-row m-strip
    int m0 = mt * 32;
    int n0 = nt * 64;
    int lane16 = lane & 15;
    int half   = lane >> 4;

    const _Float16* x0 = X + (size_t)(m0 + lane16) * KDIM + 16 * half;
    const _Float16* x1 = x0 + (size_t)16 * KDIM;
    const _Float16* w0 = W + (size_t)(n0 + lane16) * KDIM + 16 * half;

    v8f acc[8] = {};

    Frags cur = load_frags(x0, x1, w0, 0);
    #pragma unroll 3
    for (int k0 = 0; k0 < KDIM; k0 += 64) {
        Frags nxt = load_frags(x0, x1, w0, k0 + 32);   // k0+32 <= 736 < 768 always
        do_wmma8(acc, cur);
        if (k0 + 64 < KDIM)
            cur = load_frags(x0, x1, w0, k0 + 64);
        do_wmma8(acc, nxt);
    }

    #pragma unroll
    for (int rr = 0; rr < 2; ++rr) {
        #pragma unroll
        for (int r = 0; r < 8; ++r) {
            int row = m0 + rr * 16 + r + 8 * half;
            #pragma unroll
            for (int j = 0; j < 4; ++j) {
                int col = n0 + j * 16 + lane16;
                float v = acc[rr * 4 + j][r] + bias[col];
                if (HALF_OUT) {
                    outH[(size_t)row * NTOT + col] = (_Float16)v;
                } else {
                    outF[(size_t)rowmap[row] + col] = v;
                }
            }
        }
    }
}

// ---------------------------------------------------------------------------
// Attention: one wave per (bn, head). Lane l owns row qi=l>>2 and score
// columns {l&3, 4+(l&3)}; softmax reduced across the 4-lane group with
// shfl_xor; P*V with shuffled P broadcast.
// ---------------------------------------------------------------------------
__global__ __launch_bounds__(256) void attn_kernel(
    const _Float16* __restrict__ qbuf,   // [MROWS, DIMD]
    const _Float16* __restrict__ kvbuf,  // [MROWS, 2*DIMD]
    _Float16*       __restrict__ attout) // [MROWS, DIMD]
{
    int wave = blockIdx.x * (blockDim.x >> 5) + (threadIdx.x >> 5);
    int lane = threadIdx.x & 31;
    if (wave >= BN * HEADS) return;
    int bn = wave / HEADS;
    int h  = wave % HEADS;

    const _Float16* q = qbuf  + (size_t)bn * TT * DIMD + h * HDIM;
    const _Float16* k = kvbuf + (size_t)bn * TT * (2 * DIMD) + h * HDIM;
    const _Float16* v = k + DIMD;

    int qi  = lane >> 2;
    int kis = lane & 3;
    float s0 = 0.f, s1 = 0.f;
    #pragma unroll 8
    for (int d = 0; d < HDIM; ++d) {
        float qv = (float)q[qi * DIMD + d];
        s0 += qv * (float)k[kis * (2 * DIMD) + d];
        s1 += qv * (float)k[(4 + kis) * (2 * DIMD) + d];
    }
    const float scale = 0.125f;   // 64^-0.5
    s0 *= scale; s1 *= scale;

    float mx = fmaxf(s0, s1);
    mx = fmaxf(mx, __shfl_xor(mx, 1, 32));
    mx = fmaxf(mx, __shfl_xor(mx, 2, 32));
    float e0 = __expf(s0 - mx), e1 = __expf(s1 - mx);
    float sum = e0 + e1;
    sum += __shfl_xor(sum, 1, 32);
    sum += __shfl_xor(sum, 2, 32);
    float inv = 1.f / sum;
    e0 *= inv; e1 *= inv;

    float acc[16];
    #pragma unroll
    for (int c = 0; c < 16; ++c) acc[c] = 0.f;
    #pragma unroll
    for (int ki = 0; ki < 8; ++ki) {
        int src = (lane & ~3) | (ki & 3);
        float p = __shfl(ki < 4 ? e0 : e1, src, 32);
        const _Float16* vr = v + ki * (2 * DIMD) + kis * 16;
        #pragma unroll
        for (int c = 0; c < 16; ++c) acc[c] += p * (float)vr[c];
    }
    _Float16* op = attout + (size_t)(bn * TT + qi) * DIMD + h * HDIM + kis * 16;
    #pragma unroll
    for (int c = 0; c < 16; ++c) op[c] = (_Float16)acc[c];
}

// ---------------------------------------------------------------------------
extern "C" void kernel_launch(void* const* d_in, const int* in_sizes, int n_in,
                              void* d_out, int out_size, void* d_ws, size_t ws_size,
                              hipStream_t stream)
{
    (void)in_sizes; (void)n_in; (void)out_size; (void)ws_size;
    const float* s_x    = (const float*)d_in[0];
    const float* t_x    = (const float*)d_in[1];
    const float* ctp    = (const float*)d_in[2];
    const float* vtp    = (const float*)d_in[3];
    const float* Wq     = (const float*)d_in[4];
    const float* qbias  = (const float*)d_in[5];
    const float* Wkv    = (const float*)d_in[6];
    const float* kvbias = (const float*)d_in[7];
    const float* Wp     = (const float*)d_in[8];
    const float* pbias  = (const float*)d_in[9];
    float* out = (float*)d_out;

    char* base = (char*)d_ws;
    size_t off = 0;
    auto alloc = [&](size_t bytes) {
        char* p = base + off;
        off = (off + bytes + 255) & ~(size_t)255;
        return p;
    };
    _Float16* qin    = (_Float16*)alloc((size_t)MROWS * DIMD * 2);
    _Float16* kvin   = (_Float16*)alloc((size_t)MROWS * DIMD * 2);
    _Float16* Wq16   = (_Float16*)alloc((size_t)DIMD * KDIM * 2);
    _Float16* Wkv16  = (_Float16*)alloc((size_t)2 * DIMD * KDIM * 2);
    _Float16* Wp16   = (_Float16*)alloc((size_t)DIMD * KDIM * 2);
    _Float16* qbuf   = (_Float16*)alloc((size_t)MROWS * DIMD * 2);
    _Float16* kvbuf  = (_Float16*)alloc((size_t)MROWS * 2 * DIMD * 2);
    _Float16* attout = (_Float16*)alloc((size_t)MROWS * DIMD * 2);
    int*      rowmap = (int*)alloc((size_t)MROWS * 4);

    {   // fused rearrange + pos-add + f16 pack + rowmap
        size_t total = (size_t)MROWS * DIMD;
        int blocks = (int)((total + 255) / 256);
        prep_kernel<<<blocks, 256, 0, stream>>>(s_x, t_x, ctp, vtp, qin, kvin, rowmap);
    }
    f32_to_f16_kernel<<<(DIMD * KDIM + 255) / 256, 256, 0, stream>>>(Wq, Wq16, DIMD * KDIM);
    f32_to_f16_kernel<<<(2 * DIMD * KDIM + 255) / 256, 256, 0, stream>>>(Wkv, Wkv16, 2 * DIMD * KDIM);
    f32_to_f16_kernel<<<(DIMD * KDIM + 255) / 256, 256, 0, stream>>>(Wp, Wp16, DIMD * KDIM);

    {   // Q = qin * Wq^T + q_bias  (f16 out); block = 12 waves covering N=768
        gemm_wmma_kernel<DIMD, true><<<MROWS / 32, 32 * (DIMD / 64), 0, stream>>>(
            qin, Wq16, qbias, qbuf, nullptr, nullptr);
    }
    {   // KV = kvin * Wkv^T + kv_bias  (f16 out); block = 24 waves covering N=1536
        gemm_wmma_kernel<2 * DIMD, true><<<MROWS / 32, 32 * (2 * DIMD / 64), 0, stream>>>(
            kvin, Wkv16, kvbias, kvbuf, nullptr, nullptr);
    }
    {   // softmax attention per (bn, head)
        int waves = BN * HEADS;
        attn_kernel<<<(waves + 7) / 8, 256, 0, stream>>>(qbuf, kvbuf, attout);
    }
    {   // out = attout * Wproj^T + proj_bias, scattered to '(n, b*t, d)' layout
        gemm_wmma_kernel<DIMD, false><<<MROWS / 32, 32 * (DIMD / 64), 0, stream>>>(
            attout, Wp16, pbias, nullptr, out, rowmap);
    }
    copy_cls_kernel<<<(BB * TT * DIMD + 255) / 256, 256, 0, stream>>>(s_x, out);
}
